// LlamaAttention_19095424598744
// MI455X (gfx1250) — matrix operand order repaired and checked by reading, not compiled, not run on hardware
//
#include <hip/hip_runtime.h>

typedef __bf16 bf16_t;
typedef __attribute__((ext_vector_type(16))) __bf16 v16bf;
typedef __attribute__((ext_vector_type(8)))  __bf16 v8bf;
typedef __attribute__((ext_vector_type(8)))  float  v8f;

#define S_LEN 2048
#define HDIM  4096
#define NHEAD 32
#define NKVH  8
#define HEADD 128
#define FQ    (NHEAD*HEADD)      /* 4096 */
#define FKV   (NKVH*HEADD)       /* 1024 */
#define FQKV  (FQ + 2*FKV)       /* 6144 */

static __device__ __forceinline__ v8f wmma_bf16f32(v16bf a, v16bf b, v8f c) {
  // D = A(16x32 bf16) * B(32x16 bf16) + C(16x16 f32)
  return __builtin_amdgcn_wmma_f32_16x16x32_bf16(false, a, false, b, (short)0, c, false, false);
}

static __device__ __forceinline__ v8f v8f_zero() {
  v8f z;
  #pragma unroll
  for (int i = 0; i < 8; ++i) z[i] = 0.0f;
  return z;
}

// xor-shuffle within 16-lane groups via v_permlane16_b32 (pure VALU, no DS).
static constexpr unsigned pl_nib(int j, int m) { return (unsigned)((j ^ m) & 15); }
template<int M>
static constexpr unsigned pl_sel(int base) {
  return pl_nib(base + 0, M)       | pl_nib(base + 1, M) << 4  |
         pl_nib(base + 2, M) << 8  | pl_nib(base + 3, M) << 12 |
         pl_nib(base + 4, M) << 16 | pl_nib(base + 5, M) << 20 |
         pl_nib(base + 6, M) << 24 | pl_nib(base + 7, M) << 28;
}
template<int M>
static __device__ __forceinline__ float permxor16(float v) {
  const int r = __builtin_amdgcn_permlane16(
      0, __float_as_int(v), (int)pl_sel<M>(0), (int)pl_sel<M>(8), false, false);
  return __int_as_float(r);
}
static __device__ __forceinline__ float redmax16(float v) {
  v = fmaxf(v, permxor16<1>(v));
  v = fmaxf(v, permxor16<2>(v));
  v = fmaxf(v, permxor16<4>(v));
  v = fmaxf(v, permxor16<8>(v));
  return v;
}
static __device__ __forceinline__ float redsum16(float v) {
  v += permxor16<1>(v);
  v += permxor16<2>(v);
  v += permxor16<4>(v);
  v += permxor16<8>(v);
  return v;
}

// A fragment (16x32, row-major source): lane&15 = row M; K elements split as
// {kh*8+0..7, kh*8+16..23}, kh = lane>>4.  Two 16B loads.
static __device__ __forceinline__ v16bf load_fragA(const bf16_t* __restrict__ p0, int ldr) {
  const int lane = threadIdx.x & 31;
  const bf16_t* p = p0 + (size_t)(lane & 15) * ldr + (lane >> 4) * 8;
  union { v16bf v; v8bf h[2]; } u;
  u.h[0] = *(const v8bf*)(p);
  u.h[1] = *(const v8bf*)(p + 16);
  return u.v;
}

// B fragment (32x16), source stored K-contiguous per column (Bt row-major [N][K]):
// lane&15 = column N; K elements split as {kh*8+0..7, kh*8+16..23}, kh = lane>>4.
static __device__ __forceinline__ v16bf load_fragB(const bf16_t* __restrict__ p0, int ldr) {
  const int lane = threadIdx.x & 31;
  const bf16_t* p = p0 + (size_t)(lane & 15) * ldr + (lane >> 4) * 8;
  union { v16bf v; v8bf h[2]; } u;
  u.h[0] = *(const v8bf*)(p);
  u.h[1] = *(const v8bf*)(p + 16);
  return u.v;
}

// Issue one async 16B global->LDS copy (ASYNCcnt-tracked, gfx1250).
static __device__ __forceinline__ void async_ld_b128(const bf16_t* g, bf16_t* l) {
  unsigned loff = (unsigned)(size_t)l;   // generic LDS pointer: low 32 bits = LDS offset
  asm volatile("global_load_async_to_lds_b128 %0, %1, off"
               :: "v"(loff), "v"(g) : "memory");
}

// ---------------- converts ----------------
__global__ void cvt_bf16_kernel(const float* __restrict__ x, bf16_t* __restrict__ y, size_t n) {
  size_t i = (size_t)blockIdx.x * blockDim.x + threadIdx.x;
  if (i < n) y[i] = (bf16_t)x[i];
}

// y[n][k] = x[k][n]; grid = (N/256, K): coalesced reads, no integer division
__global__ void cvt_bf16_t_kernel(const float* __restrict__ x, bf16_t* __restrict__ y, int K, int N) {
  const int n = blockIdx.x * 256 + threadIdx.x;
  const int k = blockIdx.y;
  if (n < N) y[(size_t)n * K + k] = (bf16_t)x[(size_t)k * N + n];
}

// ---------------- GEMM: C[M][N] = A[M][K] * Bt[N][K]^T, f32 accum ----------------
// WG = 256 threads (8 waves, 2x4), WG tile 128x256, wave tile 64x64 (16 WMMA / K-step).
template<int N, int K>
__global__ __launch_bounds__(256) void gemm_bf16_nt(
    const bf16_t* __restrict__ A, const bf16_t* __restrict__ Bt, float* __restrict__ C)
{
  const int wave = threadIdx.x >> 5;
  const int lane = threadIdx.x & 31;
  const int row0 = blockIdx.x * 128 + (wave & 1) * 64;
  const int col0 = blockIdx.y * 256 + (wave >> 1) * 64;

  v8f acc[4][4];
  #pragma unroll
  for (int i = 0; i < 4; ++i)
    #pragma unroll
    for (int j = 0; j < 4; ++j) acc[i][j] = v8f_zero();

  for (int k0 = 0; k0 < K; k0 += 32) {
    if ((k0 & 63) == 0 && k0 + 128 < K) {   // prefetch 2 lines ahead, every other step
      #pragma unroll
      for (int i = 0; i < 4; ++i) {
        __builtin_prefetch(A  + (size_t)(row0 + i * 16 + (lane & 15)) * K + k0 + 128, 0, 3);
        __builtin_prefetch(Bt + (size_t)(col0 + i * 16 + (lane & 15)) * K + k0 + 128, 0, 3);
      }
    }
    v16bf af[4], bf[4];
    #pragma unroll
    for (int i = 0; i < 4; ++i)
      af[i] = load_fragA(A + (size_t)(row0 + i * 16) * K + k0, K);
    #pragma unroll
    for (int j = 0; j < 4; ++j)
      bf[j] = load_fragB(Bt + (size_t)(col0 + j * 16) * K + k0, K);
    #pragma unroll
    for (int i = 0; i < 4; ++i)
      #pragma unroll
      for (int j = 0; j < 4; ++j)
        acc[i][j] = wmma_bf16f32(af[i], bf[j], acc[i][j]);
  }

  const int cn = lane & 15;
  const int mh = (lane >> 4) * 8;           // C layout: lanes 16..31 hold rows r+8
  #pragma unroll
  for (int i = 0; i < 4; ++i)
    #pragma unroll
    for (int j = 0; j < 4; ++j) {
      float* Cp = C + (size_t)(row0 + i * 16 + mh) * N + (col0 + j * 16 + cn);
      #pragma unroll
      for (int r = 0; r < 8; ++r)
        Cp[r * N] = acc[i][j][r];           // constant strides -> immediate offsets
    }
}

// ---------------- RoPE + scatter to head-major Q/K and transposed V ----------------
__global__ void rope_scatter_kernel(
    const float* __restrict__ qkv, const int* __restrict__ pos,
    bf16_t* __restrict__ Qb, bf16_t* __restrict__ Kb, bf16_t* __restrict__ Vt)
{
  size_t idx = (size_t)blockIdx.x * blockDim.x + threadIdx.x;
  if (idx >= (size_t)S_LEN * FQKV) return;
  const int f = (int)(idx % FQKV);
  const int s = (int)(idx / FQKV);
  const float* row = qkv + (size_t)s * FQKV;

  if (f >= FQ + FKV) {                       // V: Vt[h][d][s]
    const int g = f - FQ - FKV;
    const int hh = g / HEADD, d = g % HEADD;
    Vt[((size_t)hh * HEADD + d) * S_LEN + s] = (bf16_t)row[f];
    return;
  }
  const bool isq = (f < FQ);
  const int g = isq ? f : (f - FQ);
  const int hh = g / HEADD, d = g % HEADD;
  const int base = isq ? hh * HEADD : FQ + hh * HEADD;
  const int dl = d & 63;
  const float p = (float)pos[s];
  const float ang = p * __powf(10000.0f, -(float)(2 * dl) * (1.0f / (float)HEADD));
  const float cs = __cosf(ang), sn = __sinf(ang);
  const float x  = row[base + d];
  const float xp = row[base + ((d < 64) ? d + 64 : d - 64)];
  const float o  = (d < 64) ? (x * cs - xp * sn) : (x * cs + xp * sn);
  if (isq) Qb[((size_t)hh * S_LEN + s) * HEADD + d] = (bf16_t)o;
  else     Kb[((size_t)hh * S_LEN + s) * HEADD + d] = (bf16_t)o;
}

// ---------------- Flash attention ----------------
// One WG = 64 q rows of one head, 4 waves (one 16-row stripe each).
// K/V blocks staged cooperatively into LDS with global_load_async_to_lds_b128.
__global__ __launch_bounds__(128) void attn_fa_kernel(
    const bf16_t* __restrict__ Qb,   // [NH][S][HD]
    const bf16_t* __restrict__ Kb,   // [NKV][S][HD]
    const bf16_t* __restrict__ Vt,   // [NKV][HD][S]
    bf16_t* __restrict__ attn)       // [S][NH*HD]
{
  const int h    = blockIdx.y;
  const int qblk = blockIdx.x;
  const int tid  = threadIdx.x;
  const int wv   = tid >> 5;
  const int lane = tid & 31;
  const int qs   = qblk * 64 + wv * 16;         // this wave's 16-row q stripe
  const int cn   = lane & 15;
  const int mh   = (lane >> 4) * 8;

  const bf16_t* Qh = Qb + (size_t)h * S_LEN * HEADD;
  const bf16_t* Kh = Kb + (size_t)(h / (NHEAD / NKVH)) * S_LEN * HEADD;
  const bf16_t* Vh = Vt + (size_t)(h / (NHEAD / NKVH)) * HEADD * S_LEN;

  __shared__ alignas(16) bf16_t Ks[32][136];    // K block: 32 keys x 128 dims (pad->conflict-free)
  __shared__ alignas(16) bf16_t Vs[128][40];    // V block: 128 dims x 32 keys
  __shared__ alignas(16) bf16_t Plds[4][16][40];// per-wave 16x32 P tile

  v16bf qf[4];
  #pragma unroll
  for (int c = 0; c < 4; ++c)
    qf[c] = load_fragA(Qh + (size_t)qs * HEADD + c * 32, HEADD);

  v8f o[8];
  #pragma unroll
  for (int t = 0; t < 8; ++t) o[t] = v8f_zero();
  float rm[8], rl[8];
  #pragma unroll
  for (int r = 0; r < 8; ++r) { rm[r] = -3.0e30f; rl[r] = 0.0f; }

  const float scale = 0.08838834764831845f;     // 1/sqrt(128)
  const int kend = qblk * 64 + 64;              // uniform causal bound for whole WG

  for (int kb = 0; kb < kend; kb += 32) {
    // ---- async stage: K 32x128 (8KB) + V 128x32 (8KB), 16B chunks, coalesced ----
    #pragma unroll
    for (int i = 0; i < 4; ++i) {
      const int c = i * 128 + tid;              // chunk id 0..511
      const int kr = c >> 4, kc = (c & 15) * 8; // K: 16 chunks per 128-elem row
      async_ld_b128(Kh + (size_t)(kb + kr) * HEADD + kc, &Ks[kr][kc]);
      const int vr = c >> 2, vc = (c & 3) * 8;  // V: 4 chunks per 32-elem row
      async_ld_b128(Vh + (size_t)vr * S_LEN + kb + vc, &Vs[vr][vc]);
    }
    asm volatile("s_wait_asynccnt 0" ::: "memory");
    __syncthreads();

    // ---- scores: S = Q * K^T (from LDS); batch loads so waits stagger ----
    v16bf kf0[4], kf1[4];
    #pragma unroll
    for (int c = 0; c < 4; ++c) kf0[c] = load_fragB(&Ks[0][c * 32], 136);
    #pragma unroll
    for (int c = 0; c < 4; ++c) kf1[c] = load_fragB(&Ks[16][c * 32], 136);
    v8f st[2];
    st[0] = v8f_zero(); st[1] = v8f_zero();
    #pragma unroll
    for (int c = 0; c < 4; ++c) st[0] = wmma_bf16f32(qf[c], kf0[c], st[0]);
    #pragma unroll
    for (int c = 0; c < 4; ++c) st[1] = wmma_bf16f32(qf[c], kf1[c], st[1]);

    float nm[8];
    #pragma unroll
    for (int r = 0; r < 8; ++r) {               // mask + scale + running row-max
      const int row = qs + mh + r;
      #pragma unroll
      for (int t = 0; t < 2; ++t) {
        const int col = kb + t * 16 + cn;
        const float v = st[t][r] * scale;
        st[t][r] = (col <= row) ? v : -3.0e30f;
      }
      const float mx = redmax16(fmaxf(st[0][r], st[1][r]));  // v_permlane16, no DS
      nm[r] = fmaxf(rm[r], mx);                 // fully masked block -> nm == rm
    }

    #pragma unroll
    for (int r = 0; r < 8; ++r) {               // exp, row-sum, rescale O
      const float alpha = __expf(rm[r] - nm[r]);
      const float p0 = __expf(st[0][r] - nm[r]);
      const float p1 = __expf(st[1][r] - nm[r]);
      st[0][r] = p0; st[1][r] = p1;
      rl[r] = rl[r] * alpha + redsum16(p0 + p1);
      rm[r] = nm[r];
      #pragma unroll
      for (int t = 0; t < 8; ++t) o[t][r] *= alpha;
    }

    // ---- P (C layout) -> LDS -> A fragment ----
    #pragma unroll
    for (int r = 0; r < 8; ++r) {
      Plds[wv][mh + r][cn]      = (bf16_t)st[0][r];
      Plds[wv][mh + r][16 + cn] = (bf16_t)st[1][r];
    }
    asm volatile("s_wait_dscnt 0" ::: "memory");
    v16bf pf = load_fragA(&Plds[wv][0][0], 40);

    // ---- O += P * V (from LDS), two 4-frag batches ----
    v16bf vfa[4], vfb[4];
    #pragma unroll
    for (int t = 0; t < 4; ++t) vfa[t] = load_fragB(&Vs[t * 16][0], 40);
    #pragma unroll
    for (int t = 0; t < 4; ++t) o[t] = wmma_bf16f32(pf, vfa[t], o[t]);
    #pragma unroll
    for (int t = 0; t < 4; ++t) vfb[t] = load_fragB(&Vs[(t + 4) * 16][0], 40);
    #pragma unroll
    for (int t = 0; t < 4; ++t) o[t + 4] = wmma_bf16f32(pf, vfb[t], o[t + 4]);

    __syncthreads();                            // protect Ks/Vs against next block's async writes
  }

  #pragma unroll
  for (int t = 0; t < 8; ++t)
    #pragma unroll
    for (int r = 0; r < 8; ++r) {
      const float inv = __builtin_amdgcn_rcpf(rl[r]);
      attn[(size_t)(qs + mh + r) * (NHEAD * HEADD) + h * HEADD + t * 16 + cn] =
          (bf16_t)(o[t][r] * inv);
    }
}

// ---------------- launcher ----------------
// workspace layout (needs ~144 MB):
//   [0,16MB)    hidden_bf  -> reused as attn_bf after GEMM1
//   [16,64MB)   wqkvT      -> reused as Qb(16MB)+Kb(4MB)+Vt(4MB) after GEMM1
//   [64,112MB)  qkv f32
//   [112,144MB) woT
extern "C" void kernel_launch(void* const* d_in, const int* in_sizes, int n_in,
                              void* d_out, int out_size, void* d_ws, size_t ws_size,
                              hipStream_t stream)
{
  (void)in_sizes; (void)n_in; (void)out_size; (void)ws_size;
  const float* hidden = (const float*)d_in[0];
  const int*   pos    = (const int*)d_in[1];
  const float* w_qkv  = (const float*)d_in[2];
  const float* w_o    = (const float*)d_in[3];
  float* out = (float*)d_out;

  char* ws = (char*)d_ws;
  const size_t MB = 1024u * 1024u;
  bf16_t* hidden_bf = (bf16_t*)(ws + 0);
  bf16_t* wqkvT     = (bf16_t*)(ws + 16 * MB);
  float*  qkv32     = (float*)(ws + 64 * MB);
  bf16_t* woT       = (bf16_t*)(ws + 112 * MB);
  bf16_t* attn_bf = hidden_bf;               // alias: hidden dead after GEMM1
  bf16_t* Qb = wqkvT;                        // alias: wqkvT dead after GEMM1
  bf16_t* Kb = (bf16_t*)(ws + 32 * MB);
  bf16_t* Vt = (bf16_t*)(ws + 36 * MB);

  { size_t n = (size_t)S_LEN * HDIM;
    cvt_bf16_kernel<<<(unsigned)((n + 255) / 256), 256, 0, stream>>>(hidden, hidden_bf, n); }
  cvt_bf16_t_kernel<<<dim3(FQKV / 256, HDIM), 256, 0, stream>>>(w_qkv, wqkvT, HDIM, FQKV);
  cvt_bf16_t_kernel<<<dim3(HDIM / 256, HDIM), 256, 0, stream>>>(w_o, woT, HDIM, HDIM);

  gemm_bf16_nt<FQKV, HDIM><<<dim3(S_LEN / 128, FQKV / 256), 256, 0, stream>>>(
      hidden_bf, wqkvT, qkv32);

  { size_t n = (size_t)S_LEN * FQKV;
    rope_scatter_kernel<<<(unsigned)((n + 255) / 256), 256, 0, stream>>>(qkv32, pos, Qb, Kb, Vt); }

  attn_fa_kernel<<<dim3(S_LEN / 64, NHEAD), 128, 0, stream>>>(Qb, Kb, Vt, attn_bf);

  gemm_bf16_nt<HDIM, HDIM><<<dim3(S_LEN / 128, HDIM / 256), 256, 0, stream>>>(
      attn_bf, woT, out);
}
